// TextGuidedSRNet_51161650430484
// MI455X (gfx1250) — compile-verified, matrix operand order checked by reading, not run on hardware
//
#include <hip/hip_runtime.h>
#include <hip/hip_bf16.h>

// ---------------------------------------------------------------------------
// TextGuidedSRNet forward for gfx1250 (MI455X), WMMA f16 everywhere.
//
//  * All matmul operands staged K-contiguous f16 -> every fragment half is one
//    b128 load (global or LDS).
//  * N-register-blocking (NB=4): one A fragment feeds 4 WMMAs (wave covers 64
//    output channels) -> 4x fewer A loads, 4x denser WMMA issue.
//  * Big convs: 16x32 A tile staged in LDS per k-step, shared by all 4 waves
//    (ds_load_b128 reads); weights stream from L2 as b128.
//  * C=64 GEMMs: 8KB weight panel staged once in LDS per block.
//  * Guards are clamped-address + cndmask selects; EXEC stays all-ones.
// ---------------------------------------------------------------------------

typedef __attribute__((ext_vector_type(16))) _Float16 v16h;
typedef __attribute__((ext_vector_type(8)))  _Float16 v8h;
typedef __attribute__((ext_vector_type(8)))  float    v8f;

#define DEV __device__ __forceinline__

DEV v8f wmma32(v16h a, v16h b, v8f c) {
  return __builtin_amdgcn_wmma_f32_16x16x32_f16(false, a, false, b,
                                                (short)0, c, false, false);
}

// K index held by element j of a v16h fragment for this lane (ISA 7.12.2).
DEV int kmap(int lane, int j) {
  return (j & 7) + ((lane >> 4) << 3) + ((j >> 3) << 4);
}

DEV v8h ld8(const _Float16* p) { return *(const v8h*)p; }

DEV v16h combine(v8h lo, v8h hi) {
  return __builtin_shufflevector(lo, hi, 0, 1, 2, 3, 4, 5, 6, 7,
                                 8, 9, 10, 11, 12, 13, 14, 15);
}

// Fragment from a K-contiguous f16 row: elements {k0+half8+0..7, +16..+23}.
DEV v16h frag_row(const _Float16* row, int k0, int half8) {
  const _Float16* p = row + k0 + half8;
  return combine(ld8(p), ld8(p + 16));
}

// ---------------------------------------------------------------------------
// Pre-conversion kernels (tiny; run once per launch)
// ---------------------------------------------------------------------------
__global__ void k_cvt(const float* __restrict__ s, _Float16* __restrict__ d, int n) {
  int i = blockIdx.x * 256 + threadIdx.x;
  if (i < n) d[i] = (_Float16)s[i];
}

// text_hidden [8,77,512] f32 -> [8,80,512] f16, pad rows zero
__global__ void k_cvt_th(const float* __restrict__ s, _Float16* __restrict__ d) {
  int i = blockIdx.x * 256 + threadIdx.x;  // over 640*512
  int row = i >> 9, k = i & 511;
  int b = row / 80, l = row % 80;
  d[i] = (l < 77) ? (_Float16)s[(b * 77 + l) * 512 + k] : (_Float16)0.f;
}

// conv weight OIHW f32 -> [CoutP][Ktot] f16, k = tap*Cin + cin, zero padded
__global__ void k_cvt_conv(const float* __restrict__ s, _Float16* __restrict__ d,
                           int Cout, int Cin) {
  int Kin = Cin * 9, Ktot = (Kin + 31) & ~31, CoutP = (Cout + 15) & ~15;
  int i = blockIdx.x * 256 + threadIdx.x;
  if (i >= CoutP * Ktot) return;
  int n = i / Ktot, k = i % Ktot;
  _Float16 v = (_Float16)0.f;
  if (n < Cout && k < Kin) {
    int tap = k / Cin, cin = k % Cin;
    v = (_Float16)s[(n * Cin + cin) * 9 + tap];
  }
  d[i] = v;
}

__global__ void k_zero_h(_Float16* __restrict__ d, int n) {
  int i = blockIdx.x * 256 + threadIdx.x;
  if (i < n) d[i] = (_Float16)0.f;
}

// ---------------------------------------------------------------------------
// 1) Text projection: tf_h[m,c] = thH[m,:] . pwH[c,:] + pb.  M=640,N=64,K=512.
//    (small; direct loads, waves split N)
// ---------------------------------------------------------------------------
__global__ void k_text_proj(const _Float16* __restrict__ thH,
                            const _Float16* __restrict__ pwH,
                            const float* __restrict__ pb,
                            _Float16* __restrict__ tfH) {
  const int lane = threadIdx.x & 31;
  const int wave = threadIdx.x >> 5;
  const int rc = lane & 15;
  const int half8 = (lane >> 4) << 3;
  const int m0 = blockIdx.x * 16;
  const int n0 = wave * 16;
  const _Float16* ra = thH + (m0 + rc) * 512;
  const _Float16* rb = pwH + (n0 + rc) * 512;
  v8f acc = {};
  for (int k0 = 0; k0 < 512; k0 += 32)
    acc = wmma32(frag_row(ra, k0, half8), frag_row(rb, k0, half8), acc);
#pragma unroll
  for (int i = 0; i < 8; ++i) {
    int mm = m0 + half8 + i;
    int nn = n0 + rc;
    float v = (mm % 80 < 77) ? (acc[i] + pb[nn]) : 0.f;
    tfH[mm * 64 + nn] = (_Float16)v;
  }
}

// ---------------------------------------------------------------------------
// 2) First conv 3->64 (x NCHW f32, scalar A gather, K=27 pad 32 tap-major).
//    Writes feat master f32 NHWC + f16 mirror.
// ---------------------------------------------------------------------------
__global__ void k_conv_cf(const float* __restrict__ x,
                          const _Float16* __restrict__ w,  // [64][32]
                          const float* __restrict__ bias,
                          float* __restrict__ featF, _Float16* __restrict__ featH) {
  const int lane = threadIdx.x & 31;
  const int wave = threadIdx.x >> 5;
  const int rc = lane & 15;
  const int half8 = (lane >> 4) << 3;
  const int m0 = blockIdx.x * 16;
  const int n0 = wave * 16;
  const int m = m0 + rc;
  const int b = m >> 12, p = m & 4095;
  const int y = p >> 6, xq = p & 63;
  v16h a;
#pragma unroll
  for (int j = 0; j < 16; ++j) {
    int k = kmap(lane, j);
    int tap = k / 3, cin = k - tap * 3;
    int yy = y + tap / 3 - 1, xx = xq + tap % 3 - 1;
    bool ok = (k < 27) & ((unsigned)yy < 64u) & ((unsigned)xx < 64u);
    int yc = ok ? yy : y, xc = ok ? xx : xq;
    float av = x[(((b * 3 + cin) << 6) + yc) * 64 + xc] * (ok ? 1.f : 0.f);
    a[j] = (_Float16)av;
  }
  v16h bm = frag_row(w + (n0 + rc) * 32, 0, half8);
  v8f acc = {};
  acc = wmma32(a, bm, acc);
#pragma unroll
  for (int i = 0; i < 8; ++i) {
    int mm = m0 + half8 + i;
    int nn = n0 + rc;
    float v = acc[i] + bias[nn];
    featF[mm * 64 + nn] = v;
    featH[mm * 64 + nn] = (_Float16)v;
  }
}

// ---------------------------------------------------------------------------
// 3) Big 3x3 conv (Cin=64, Cout=256), NHWC f16, implicit GEMM, K=576.
//    A tile (16x32) staged in LDS per k-step, shared by all 4 waves; each wave
//    computes 16M x 64N (NB=4).  Fused pixel-shuffle(2)+ReLU f16 NHWC out.
// ---------------------------------------------------------------------------
DEV v8h conv_a_group(const _Float16* __restrict__ in, int b, int y, int x,
                     int Hh, int Ww, int kk) {
  int tap = kk >> 6;                 // Cin = 64
  int cin = kk & 63;
  int yy = y + tap / 3 - 1;
  int xx = x + tap % 3 - 1;
  bool inb = ((unsigned)yy < (unsigned)Hh) & ((unsigned)xx < (unsigned)Ww);
  int yc = inb ? yy : y, xc = inb ? xx : x;
  v8h v = ld8(in + ((((long)b * Hh + yc) * Ww + xc) << 6) + cin);
  v8h z = {};
  return inb ? v : z;
}

__global__ void k_conv3h_big(const _Float16* __restrict__ in,
                             const _Float16* __restrict__ w,   // [256][576]
                             const float* __restrict__ bias,
                             _Float16* __restrict__ out,       // NHWC [B,2H,2W,64]
                             int Hh, int Ww) {
  __shared__ _Float16 la[16 * 32];
  const int tid = threadIdx.x;
  const int lane = tid & 31;
  const int wave = tid >> 5;
  const int rc = lane & 15;
  const int half8 = (lane >> 4) << 3;
  const int m0 = blockIdx.x * 16;
  const int HW = Hh * Ww;
  // staging thread coords (threads 0..63 stage; unit = 8 consecutive k)
  const int sr = tid >> 2, sg = tid & 3;
  const int sm = m0 + sr;
  const int sb = sm / HW;
  const int srm = sm % HW;
  const int sy = srm / Ww, sx = srm % Ww;
  v8f acc[4] = {};
  for (int k0 = 0; k0 < 576; k0 += 32) {
    __syncthreads();
    if (tid < 64) {
      v8h vv = conv_a_group(in, sb, sy, sx, Hh, Ww, k0 + sg * 8);
      *(v8h*)(la + sr * 32 + sg * 8) = vv;
    }
    __syncthreads();
    v16h a = combine(*(const v8h*)(la + rc * 32 + half8),
                     *(const v8h*)(la + rc * 32 + 16 + half8));
#pragma unroll
    for (int t = 0; t < 4; ++t) {
      v16h bm = frag_row(w + (long)(wave * 64 + t * 16 + rc) * 576, k0, half8);
      acc[t] = wmma32(a, bm, acc[t]);
    }
  }
#pragma unroll
  for (int t = 0; t < 4; ++t) {
#pragma unroll
    for (int i = 0; i < 8; ++i) {
      int nn = wave * 64 + t * 16 + rc;
      int mm = m0 + half8 + i;
      int bb = mm / HW;
      int rr = mm % HW;
      int yy = rr / Ww, xx = rr % Ww;
      float v = acc[t][i] + bias[nn];
      int c = nn >> 2, r1 = (nn >> 1) & 1, r2 = nn & 1;
      long o = ((((long)bb * (Hh * 2) + (yy * 2 + r1)) * (Ww * 2)) +
                (xx * 2 + r2)) * 64 + c;
      out[o] = (_Float16)(v > 0.f ? v : 0.f);
    }
  }
}

// ---------------------------------------------------------------------------
// 4) Final 3x3 conv 64->3, NHWC f16 in, NCHW f32 out (wave 0 only active).
// ---------------------------------------------------------------------------
__global__ void k_conv3h_fin(const _Float16* __restrict__ in,
                             const _Float16* __restrict__ w,   // [16][576]
                             const float* __restrict__ bias,
                             float* __restrict__ out,
                             int Hh, int Ww) {
  const int lane = threadIdx.x & 31;
  const int wave = threadIdx.x >> 5;
  if (wave != 0) return;  // uniform whole-wave exit
  const int rc = lane & 15;
  const int half8 = (lane >> 4) << 3;
  const int m0 = blockIdx.x * 16;
  const int HW = Hh * Ww;
  const int m = m0 + rc;
  const int b = m / HW;
  const int rem = m % HW;
  const int y = rem / Ww, x = rem % Ww;
  const _Float16* rb = w + (long)rc * 576;
  v8f acc = {};
  for (int k0 = 0; k0 < 576; k0 += 32) {
    v8h alo = conv_a_group(in, b, y, x, Hh, Ww, k0 + half8);
    v8h ahi = conv_a_group(in, b, y, x, Hh, Ww, k0 + 16 + half8);
    acc = wmma32(combine(alo, ahi), frag_row(rb, k0, half8), acc);
  }
#pragma unroll
  for (int i = 0; i < 8; ++i) {
    int nn = rc;
    if (nn >= 3) continue;
    int mm = m0 + half8 + i;
    int bb = mm / HW;
    int rr = mm % HW;
    int yy = rr / Ww, xx = rr % Ww;
    out[(((long)bb * 3 + nn) * Hh + yy) * Ww + xx] = acc[i] + bias[nn];
  }
}

// ---------------------------------------------------------------------------
// Shared helper: C=64 GEMM with 8KB weight panel in LDS, NB=4 per wave.
// ---------------------------------------------------------------------------
DEV void stage_w64(_Float16* lb, const _Float16* __restrict__ wsrc, int tid) {
  int t4 = tid * 8;
#pragma unroll
  for (int i = 0; i < 4; ++i)
    *(v8h*)(lb + t4 + i * 1024) = ld8(wsrc + t4 + i * 1024);
}

DEV void gemm64_nb4(const _Float16* __restrict__ aRow, const _Float16* lb,
                    int half8, int rc, v8f acc[4]) {
#pragma unroll
  for (int k0 = 0; k0 < 64; k0 += 32) {
    v16h a = frag_row(aRow, k0, half8);
#pragma unroll
    for (int t = 0; t < 4; ++t) {
      v16h b = frag_row(lb + (t * 16 + rc) * 64, k0, half8);
      acc[t] = wmma32(a, b, acc[t]);
    }
  }
}

// ---------------------------------------------------------------------------
// 5) Q projection + positional encoding.  Block = 64M x 64N.  f32 out for LN.
// ---------------------------------------------------------------------------
__global__ void k_qproj(const _Float16* __restrict__ featH,
                        const _Float16* __restrict__ qwH,
                        const float* __restrict__ qb,
                        float* __restrict__ qout) {
  __shared__ _Float16 lb[4096];
  const int tid = threadIdx.x;
  stage_w64(lb, qwH, tid);
  __syncthreads();
  const int lane = tid & 31;
  const int wave = tid >> 5;
  const int rc = lane & 15;
  const int half8 = (lane >> 4) << 3;
  const int m0 = blockIdx.x * 64 + wave * 16;
  v8f acc[4] = {};
  gemm64_nb4(featH + (m0 + rc) * 64, lb, half8, rc, acc);
#pragma unroll
  for (int t = 0; t < 4; ++t) {
#pragma unroll
    for (int i = 0; i < 8; ++i) {
      int mm = m0 + half8 + i;
      int nn = t * 16 + rc;
      int pp = mm & 4095;
      int yy = pp >> 6, xx = pp & 63;
      float pe = (nn < 32) ? (float)xx * (1.f / 63.f) : (float)yy * (1.f / 63.f);
      qout[mm * 64 + nn] = acc[t][i] + qb[nn] + 0.05f * pe;
    }
  }
}

// ---------------------------------------------------------------------------
// 6) K/V projections.  blockIdx.y: 0 -> K (f32 rows for LN), 1 -> V (f16,
//    transposed to vT[b*64+c][96], rows l>=77 zeroed).  M=640.
// ---------------------------------------------------------------------------
__global__ void k_kvproj(const _Float16* __restrict__ tfH,
                         const _Float16* __restrict__ kwH, const float* __restrict__ kb,
                         const _Float16* __restrict__ vwH, const float* __restrict__ vb,
                         float* __restrict__ kout, _Float16* __restrict__ vT) {
  __shared__ _Float16 lb[4096];
  const int tid = threadIdx.x;
  const bool isV = blockIdx.y != 0;
  stage_w64(lb, isV ? vwH : kwH, tid);
  __syncthreads();
  const int lane = tid & 31;
  const int wave = tid >> 5;
  const int rc = lane & 15;
  const int half8 = (lane >> 4) << 3;
  const int m0 = blockIdx.x * 64 + wave * 16;
  const float* bb = isV ? vb : kb;
  v8f acc[4] = {};
  gemm64_nb4(tfH + (m0 + rc) * 64, lb, half8, rc, acc);
#pragma unroll
  for (int t = 0; t < 4; ++t) {
#pragma unroll
    for (int i = 0; i < 8; ++i) {
      int mm = m0 + half8 + i;        // padded row index (b*80 + l)
      int nn = t * 16 + rc;
      int bq = mm / 80, ll = mm % 80;
      float v = acc[t][i] + bb[nn];
      if (isV) vT[(bq * 64 + nn) * 96 + ll] = (_Float16)((ll < 77) ? v : 0.f);
      else     kout[mm * 64 + nn] = v;
    }
  }
}

// ---------------------------------------------------------------------------
// 7) LayerNorm over rows of 64 (one wave/row, shuffle reductions), f16 out.
// ---------------------------------------------------------------------------
__global__ void k_ln64(const float* __restrict__ in,
                       const float* __restrict__ w, const float* __restrict__ b,
                       _Float16* __restrict__ out, int R) {
  int row = blockIdx.x * (blockDim.x >> 5) + (threadIdx.x >> 5);
  if (row >= R) return;
  int lane = threadIdx.x & 31;
  const float* p = in + (long)row * 64;
  float v0 = p[lane], v1 = p[lane + 32];
  float s = v0 + v1;
#pragma unroll
  for (int o = 16; o; o >>= 1) s += __shfl_xor(s, o, 32);
  float mean = s * (1.f / 64.f);
  float d0 = v0 - mean, d1 = v1 - mean;
  float q = d0 * d0 + d1 * d1;
#pragma unroll
  for (int o = 16; o; o >>= 1) q += __shfl_xor(q, o, 32);
  float rinv = rsqrtf(q * (1.f / 64.f) + 1e-5f);
  out[row * 64 + lane]      = (_Float16)(d0 * rinv * w[lane] + b[lane]);
  out[row * 64 + lane + 32] = (_Float16)(d1 * rinv * w[lane + 32] + b[lane + 32]);
}

// ---------------------------------------------------------------------------
// 8) Scores: S[bh,m,l] = scale * q[b,m,h*8:+8] . kn[b,l,h*8:+8].
// ---------------------------------------------------------------------------
__global__ void k_scores(const _Float16* __restrict__ qn,
                         const _Float16* __restrict__ kn,
                         float* __restrict__ sc) {
  const int lane = threadIdx.x & 31;
  const int wave = threadIdx.x >> 5;
  const int rc = lane & 15;
  const int half = lane >> 4;
  const int half8 = half << 3;
  const int bh = blockIdx.z;
  const int b = bh >> 3, h = bh & 7;
  const int m0 = blockIdx.x * 64 + wave * 16;
  const int n0 = blockIdx.y * 16;
  v8h z = {};
  v8h qa = ld8(qn + ((long)(b << 12) + m0 + rc) * 64 + h * 8);
  v8h kb = ld8(kn + ((long)b * 80 + n0 + rc) * 64 + h * 8);
  qa = half ? z : qa;
  kb = half ? z : kb;
  v8f acc = {};
  acc = wmma32(combine(qa, z), combine(kb, z), acc);
  const float scale = 0.35355339059327373f;  // 8^-0.5
#pragma unroll
  for (int i = 0; i < 8; ++i)
    sc[((long)(bh << 12) + m0 + half8 + i) * 80 + n0 + rc] = acc[i] * scale;
}

// ---------------------------------------------------------------------------
// 9) Softmax over L=77 (stride 80 in, 96 out zero-padded), f16 probs.
// ---------------------------------------------------------------------------
__global__ void k_softmax(const float* __restrict__ sc,
                          _Float16* __restrict__ p) {
  int row = blockIdx.x * (blockDim.x >> 5) + (threadIdx.x >> 5);
  int lane = threadIdx.x & 31;
  const float* s = sc + (long)row * 80;
  float x0 = s[lane];
  float x1 = s[lane + 32];
  float x2 = (lane + 64 < 77) ? s[lane + 64] : -3.4e38f;
  float mx = fmaxf(x0, fmaxf(x1, x2));
#pragma unroll
  for (int o = 16; o; o >>= 1) mx = fmaxf(mx, __shfl_xor(mx, o, 32));
  float e0 = __expf(x0 - mx);
  float e1 = __expf(x1 - mx);
  float e2 = (lane + 64 < 77) ? __expf(x2 - mx) : 0.f;
  float sum = e0 + e1 + e2;
#pragma unroll
  for (int o = 16; o; o >>= 1) sum += __shfl_xor(sum, o, 32);
  float r = 1.f / sum;
  _Float16* q = p + (long)row * 96;
  q[lane]      = (_Float16)(e0 * r);
  q[lane + 32] = (_Float16)(e1 * r);
  q[lane + 64] = (_Float16)(e2 * r);
}

// ---------------------------------------------------------------------------
// 10) AV: aout[b,m,h*8+d] = sum_l P[bh,m,l] * vT[b*64+h*8+d][l].  K=96.
// ---------------------------------------------------------------------------
__global__ void k_av(const _Float16* __restrict__ p,
                     const _Float16* __restrict__ vT,
                     _Float16* __restrict__ aout) {
  const int lane = threadIdx.x & 31;
  const int wave = threadIdx.x >> 5;
  const int rc = lane & 15;
  const int half8 = (lane >> 4) << 3;
  const int bh = blockIdx.y;
  const int b = bh >> 3, h = bh & 7;
  const int m0 = blockIdx.x * 64 + wave * 16;
  const _Float16* ra = p + ((long)(bh << 12) + m0 + rc) * 96;
  const _Float16* rb = vT + ((long)(b * 64) + h * 8 + (rc & 7)) * 96;
  const bool nok = rc < 8;
  v8h z = {};
  v8f acc = {};
#pragma unroll
  for (int k0 = 0; k0 < 96; k0 += 32) {
    v8h blo = ld8(rb + k0 + half8);
    v8h bhi = ld8(rb + k0 + 16 + half8);
    blo = nok ? blo : z;
    bhi = nok ? bhi : z;
    acc = wmma32(frag_row(ra, k0, half8), combine(blo, bhi), acc);
  }
#pragma unroll
  for (int i = 0; i < 8; ++i) {
    if (rc < 8) {
      int mm = m0 + half8 + i;
      aout[((long)(b << 12) + mm) * 64 + h * 8 + rc] = (_Float16)acc[i];
    }
  }
}

// ---------------------------------------------------------------------------
// 11) Output projection + residual (NHWC).  Block = 64M x 64N.
// ---------------------------------------------------------------------------
__global__ void k_oproj(const _Float16* __restrict__ aout,
                        const _Float16* __restrict__ owH, const float* __restrict__ ob,
                        const float* __restrict__ finF,
                        float* __restrict__ foutF, _Float16* __restrict__ foutH) {
  __shared__ _Float16 lb[4096];
  const int tid = threadIdx.x;
  stage_w64(lb, owH, tid);
  __syncthreads();
  const int lane = tid & 31;
  const int wave = tid >> 5;
  const int rc = lane & 15;
  const int half8 = (lane >> 4) << 3;
  const int m0 = blockIdx.x * 64 + wave * 16;
  v8f acc[4] = {};
  gemm64_nb4(aout + (m0 + rc) * 64, lb, half8, rc, acc);
#pragma unroll
  for (int t = 0; t < 4; ++t) {
#pragma unroll
    for (int i = 0; i < 8; ++i) {
      int nn = t * 16 + rc;
      int idx = (m0 + half8 + i) * 64 + nn;
      float v = acc[t][i] + ob[nn] + finF[idx];
      foutF[idx] = v;
      foutH[idx] = (_Float16)v;
    }
  }
}

// ---------------------------------------------------------------------------
// Host side
// ---------------------------------------------------------------------------
extern "C" void kernel_launch(void* const* d_in, const int* in_sizes, int n_in,
                              void* d_out, int out_size, void* d_ws, size_t ws_size,
                              hipStream_t stream) {
  (void)in_sizes; (void)n_in; (void)out_size; (void)ws_size;
  const float* x   = (const float*)d_in[0];
  const float* th  = (const float*)d_in[1];
  const float* pw  = (const float*)d_in[2];
  const float* pb  = (const float*)d_in[3];
  const float* cfw = (const float*)d_in[4];
  const float* cfb = (const float*)d_in[5];
  const float* qw  = (const float*)d_in[6];
  const float* qb  = (const float*)d_in[7];
  const float* kw  = (const float*)d_in[8];
  const float* kb  = (const float*)d_in[9];
  const float* vw  = (const float*)d_in[10];
  const float* vb  = (const float*)d_in[11];
  const float* ow  = (const float*)d_in[12];
  const float* ob  = (const float*)d_in[13];
  const float* l1w = (const float*)d_in[14];
  const float* l1b = (const float*)d_in[15];
  const float* l2w = (const float*)d_in[16];
  const float* l2b = (const float*)d_in[17];
  const float* upw = (const float*)d_in[18];
  const float* upb = (const float*)d_in[19];
  const float* clw = (const float*)d_in[20];
  const float* clb = (const float*)d_in[21];

  char* ws = (char*)d_ws;
  size_t off = 0;
  auto alloc = [&](size_t bytes) -> void* {
    void* ptr = ws + off;
    off = (off + bytes + 255) & ~(size_t)255;
    return ptr;
  };
  _Float16* thH   = (_Float16*)alloc((size_t)640 * 512 * 2);
  _Float16* pwH   = (_Float16*)alloc((size_t)64 * 512 * 2);
  _Float16* qwH   = (_Float16*)alloc((size_t)16 * 4096 * 2);
  _Float16* kwH   = (_Float16*)alloc((size_t)16 * 4096 * 2);
  _Float16* vwH   = (_Float16*)alloc((size_t)16 * 4096 * 2);
  _Float16* owH   = (_Float16*)alloc((size_t)16 * 4096 * 2);
  _Float16* wcfH  = (_Float16*)alloc((size_t)64 * 32 * 2);
  _Float16* wup1H = (_Float16*)alloc((size_t)256 * 576 * 2);
  _Float16* wup2H = (_Float16*)alloc((size_t)256 * 576 * 2);
  _Float16* wclH  = (_Float16*)alloc((size_t)16 * 576 * 2);
  _Float16* tfH   = (_Float16*)alloc((size_t)640 * 64 * 2);
  float*    featFA= (float*)alloc((size_t)8 * 4096 * 64 * 4);
  float*    featFB= (float*)alloc((size_t)8 * 4096 * 64 * 4);
  _Float16* featHA= (_Float16*)alloc((size_t)8 * 4096 * 64 * 2);
  _Float16* featHB= (_Float16*)alloc((size_t)8 * 4096 * 64 * 2);
  float*    qbuf  = (float*)alloc((size_t)8 * 4096 * 64 * 4);
  _Float16* qn    = (_Float16*)alloc((size_t)8 * 4096 * 64 * 2);
  float*    kbuf  = (float*)alloc((size_t)640 * 64 * 4);
  _Float16* knb   = (_Float16*)alloc((size_t)640 * 64 * 2);
  _Float16* vT    = (_Float16*)alloc((size_t)512 * 96 * 2);
  float*    sc    = (float*)alloc((size_t)64 * 4096 * 80 * 4);
  _Float16* pbuf  = (_Float16*)alloc((size_t)64 * 4096 * 96 * 2);
  _Float16* aout  = (_Float16*)alloc((size_t)8 * 4096 * 64 * 2);
  _Float16* up1H  = (_Float16*)alloc((size_t)8 * 128 * 128 * 64 * 2);
  _Float16* up2H  = (_Float16*)alloc((size_t)8 * 256 * 256 * 64 * 2);

  dim3 blk(128);

  // ---- one-time staging: f16 weights / padded text ----
  k_cvt_th<<<640 * 512 / 256, 256, 0, stream>>>(th, thH);
  k_cvt<<<(64 * 512) / 256, 256, 0, stream>>>(pw, pwH, 64 * 512);
  k_cvt<<<(16 * 4096) / 256, 256, 0, stream>>>(qw, qwH, 16 * 4096);
  k_cvt<<<(16 * 4096) / 256, 256, 0, stream>>>(kw, kwH, 16 * 4096);
  k_cvt<<<(16 * 4096) / 256, 256, 0, stream>>>(vw, vwH, 16 * 4096);
  k_cvt<<<(16 * 4096) / 256, 256, 0, stream>>>(ow, owH, 16 * 4096);
  k_cvt_conv<<<(64 * 32 + 255) / 256, 256, 0, stream>>>(cfw, wcfH, 64, 3);
  k_cvt_conv<<<(256 * 576 + 255) / 256, 256, 0, stream>>>(upw, wup1H, 256, 64);
  k_cvt_conv<<<(256 * 576 + 255) / 256, 256, 0, stream>>>(upw + 256 * 64 * 9, wup2H, 256, 64);
  k_cvt_conv<<<(16 * 576 + 255) / 256, 256, 0, stream>>>(clw, wclH, 3, 64);
  k_zero_h<<<(512 * 96 + 255) / 256, 256, 0, stream>>>(vT, 512 * 96);

  // ---- text projection + first conv ----
  k_text_proj<<<40, blk, 0, stream>>>(thH, pwH, pb, tfH);
  k_conv_cf<<<2048, blk, 0, stream>>>(x, wcfH, cfb, featFA, featHA);

  float*    finF = featFA;  _Float16* finH = featHA;
  float*    foF  = featFB;  _Float16* foH  = featHB;
  for (int i = 0; i < 16; ++i) {
    k_qproj<<<512, blk, 0, stream>>>(finH, qwH + i * 4096, qb + i * 64, qbuf);
    k_ln64<<<32768 / 4, blk, 0, stream>>>(qbuf, l1w + i * 64, l1b + i * 64, qn, 32768);
    k_kvproj<<<dim3(10, 2), blk, 0, stream>>>(tfH, kwH + i * 4096, kb + i * 64,
                                              vwH + i * 4096, vb + i * 64, kbuf, vT);
    k_ln64<<<160, blk, 0, stream>>>(kbuf, l2w + i * 64, l2b + i * 64, knb, 640);
    k_scores<<<dim3(64, 5, 64), blk, 0, stream>>>(qn, knb, sc);
    k_softmax<<<262144 / 4, blk, 0, stream>>>(sc, pbuf);
    k_av<<<dim3(64, 64), blk, 0, stream>>>(pbuf, vT, aout);
    k_oproj<<<512, blk, 0, stream>>>(aout, owH + i * 4096, ob + i * 64, finF, foF, foH);
    { float* t = finF; finF = foF; foF = t; }
    { _Float16* t = finH; finH = foH; foH = t; }
  }

  // ---- upsampling + final conv ----
  k_conv3h_big<<<2048, blk, 0, stream>>>(finH, wup1H, upb, up1H, 64, 64);
  k_conv3h_big<<<8192, blk, 0, stream>>>(up1H, wup2H, upb + 256, up2H, 128, 128);
  k_conv3h_fin<<<32768, blk, 0, stream>>>(up2H, wclH, clb, (float*)d_out, 256, 256);
}
